// Self_Attention_16028817949071
// MI455X (gfx1250) — matrix operand order repaired and checked by reading, not compiled, not run on hardware
//
#include <hip/hip_runtime.h>
#include <hip/hip_bf16.h>

// ---------------------------------------------------------------------------
// Types for CDNA5 WMMA (wave32): v_wmma_f32_16x16x32_bf16
// ---------------------------------------------------------------------------
typedef __attribute__((ext_vector_type(16))) __bf16 v16bf;
typedef __attribute__((ext_vector_type(8)))  __bf16 v8bf;
typedef __attribute__((ext_vector_type(8)))  float  v8f;
typedef int i32x4 __attribute__((vector_size(16)));

#define WMMA_BF16(a, b, c) \
  __builtin_amdgcn_wmma_f32_16x16x32_bf16(false, (a), false, (b), (short)0, (c), false, false)

// ---------------------------------------------------------------------------
// Async global -> LDS copy (CDNA5, tracked by ASYNCcnt).  The builtin is
// declared on this toolchain with signature
//   (int4 addrspace(1)*, int4 addrspace(3)*, imm offset, imm cpol)
// per the round-2 diagnostic (AS(1) prints as __device__).
// ---------------------------------------------------------------------------
__device__ __forceinline__ void async_copy16(__bf16* lds_dst, const __bf16* gsrc) {
#if __has_builtin(__builtin_amdgcn_global_load_async_to_lds_b128)
  __builtin_amdgcn_global_load_async_to_lds_b128(
      (__attribute__((address_space(1))) i32x4*)(void*)gsrc,
      (__attribute__((address_space(3))) i32x4*)(void*)lds_dst,
      0, 0);
#else
  asm volatile("global_load_async_to_lds_b128 %0, %1, off"
               :: "v"((unsigned)(unsigned long long)lds_dst),
                  "v"((unsigned long long)gsrc)
               : "memory");
#endif
}

__device__ __forceinline__ void wait_async_le4() {
#if __has_builtin(__builtin_amdgcn_s_wait_asynccnt)
  __builtin_amdgcn_s_wait_asynccnt(4);
#else
  asm volatile("s_wait_asynccnt 0x4" ::: "memory");
#endif
}
__device__ __forceinline__ void wait_async_0() {
#if __has_builtin(__builtin_amdgcn_s_wait_asynccnt)
  __builtin_amdgcn_s_wait_asynccnt(0);
#else
  asm volatile("s_wait_asynccnt 0x0" ::: "memory");
#endif
}

// combine two 8-element halves into a 16-element A/B operand
__device__ __forceinline__ v16bf combine16(v8bf lo, v8bf hi) {
  v16bf r;
#pragma unroll
  for (int i = 0; i < 8; ++i) { r[i] = lo[i]; r[i + 8] = hi[i]; }
  return r;
}

// B operand: 16 contiguous bf16 (32B) -> two b128 loads (global or LDS)
__device__ __forceinline__ v16bf load_b_tile(const __bf16* p) {
  return combine16(*(const v8bf*)p, *(const v8bf*)(p + 8));
}

// A operand (16-bit A layout, 16x32): per lane two contiguous 8-elt segments
__device__ __forceinline__ v16bf load_a_bf16(const __bf16* base) {
  return combine16(*(const v8bf*)base, *(const v8bf*)(base + 16));
}

__device__ __forceinline__ v16bf load_a_f32(const float* base) {
  v16bf r;
#pragma unroll
  for (int j = 0; j < 8; ++j) {
    r[j]     = (__bf16)base[j];
    r[j + 8] = (__bf16)base[16 + j];
  }
  return r;
}

// xor-tree reductions over a 16-lane half (rows of the WMMA C layout live in
// lanes 0-15 (M=0..7) and lanes 16-31 (M=8..15); xor masks 1,2,4,8 stay in-half)
__device__ __forceinline__ float halfmax(float v) {
#pragma unroll
  for (int m = 1; m < 16; m <<= 1) v = fmaxf(v, __shfl_xor(v, m, 32));
  return v;
}
__device__ __forceinline__ float halfsum(float v) {
#pragma unroll
  for (int m = 1; m < 16; m <<= 1) v += __shfl_xor(v, m, 32);
  return v;
}

// ---------------------------------------------------------------------------
// Kernel 0: repack a 64x64 f32 weight (row-major [k][n]) into bf16 in exact
// WMMA-B lane order: out[((nt*2+kt)*32 + lane)*16 + j]
// ---------------------------------------------------------------------------
__global__ void pack_weight_kernel(const float* __restrict__ w,
                                   __bf16* __restrict__ out) {
  for (int i = threadIdx.x; i < 4096; i += blockDim.x) {
    int j    = i & 15;
    int lane = (i >> 4) & 31;
    int kt   = (i >> 9) & 1;
    int nt   = i >> 10;
    int n = nt * 16 + (lane & 15);
    int K = (lane < 16) ? ((j < 8) ? j : j + 8) : ((j < 8) ? j + 8 : j + 16);
    int k = kt * 32 + K;
    out[i] = (__bf16)w[k * 64 + n];
  }
}

// ---------------------------------------------------------------------------
// Kernel 1: 1x1 conv as GEMM  out[m, 0..63] = A[m, 0..63] @ W + bias
// ---------------------------------------------------------------------------
__global__ __launch_bounds__(128)
void conv1x1_wmma_kernel(const float*  __restrict__ Af32,
                         const __bf16* __restrict__ Abf,
                         const __bf16* __restrict__ Wp,
                         const float*  __restrict__ bias,
                         __bf16*       __restrict__ out_bf,
                         float*        __restrict__ out_f32,
                         const float*  __restrict__ residual,
                         const float*  __restrict__ sigma,
                         int M) {
  const int wave = threadIdx.x >> 5;
  const int lane = threadIdx.x & 31;
  const int m0 = (blockIdx.x * (blockDim.x >> 5) + wave) * 16;
  if (m0 >= M) return;

  const int r     = lane & 15;
  const int hiofs = (lane & 16) ? 8 : 0;   // lane>=16 -> K segments shift by 8

  v16bf a[2];
#pragma unroll
  for (int kt = 0; kt < 2; ++kt) {
    long base = (long)(m0 + r) * 64 + kt * 32 + hiofs;
    a[kt] = Af32 ? load_a_f32(Af32 + base) : load_a_bf16(Abf + base);
  }

  v8f acc[4];
#pragma unroll
  for (int nt = 0; nt < 4; ++nt) {
    float b = bias[nt * 16 + r];
#pragma unroll
    for (int v = 0; v < 8; ++v) acc[nt][v] = b;
  }

#pragma unroll
  for (int nt = 0; nt < 4; ++nt) {
#pragma unroll
    for (int kt = 0; kt < 2; ++kt) {
      const __bf16* bp = Wp + ((nt * 2 + kt) * 32 + lane) * 16;
      acc[nt] = WMMA_BF16(a[kt], load_b_tile(bp), acc[nt]);
    }
  }

  const int rowofs = (lane & 16) ? 8 : 0;   // C/D layout: lanes>=16 hold M=8..15
  const float sg = out_f32 ? sigma[0] : 0.0f;
#pragma unroll
  for (int v = 0; v < 8; ++v) {
    long row = m0 + v + rowofs;
#pragma unroll
    for (int nt = 0; nt < 4; ++nt) {
      long idx = row * 64 + nt * 16 + r;
      if (out_f32) out_f32[idx] = residual[idx] + sg * acc[nt][v];
      else         out_bf[idx]  = (__bf16)acc[nt][v];
    }
  }
}

// ---------------------------------------------------------------------------
// Kernel 2: 2x2 maxpool on bf16 [n,64,64,64] -> 1024 keys/image.
//   transpose==0: out[n][key][c]  (phi)    transpose==1: out[n][c][key] (g^T)
// ---------------------------------------------------------------------------
__global__ void pool_kernel(const __bf16* __restrict__ in,
                            __bf16* __restrict__ out, int transpose) {
  int idx = blockIdx.x * blockDim.x + threadIdx.x;   // 8*32*32*64 total
  int c  = idx & 63;  int t = idx >> 6;
  int pw = t & 31;    t >>= 5;
  int ph = t & 31;    int n = t >> 5;
  long base = ((long)n * 4096 + (2 * ph) * 64 + 2 * pw) * 64 + c;
  float v0 = (float)in[base];
  float v1 = (float)in[base + 64];
  float v2 = (float)in[base + 4096];
  float v3 = (float)in[base + 4096 + 64];
  float m = fmaxf(fmaxf(v0, v1), fmaxf(v2, v3));
  int key = ph * 32 + pw;
  if (transpose) out[((long)n * 64 + c) * 1024 + key] = (__bf16)m;
  else           out[((long)n * 1024 + key) * 64 + c] = (__bf16)m;
}

// ---------------------------------------------------------------------------
// Kernel 3: flash attention with async double-buffered K/V staging in LDS.
//   Block = 4 waves = 4 consecutive 16-query tiles of the SAME batch, so all
//   waves share the K/V stream.  Each 32-key chunk (phi[32][64] + g^T[64][32],
//   8KB) is staged once per block via GLOBAL_LOAD_ASYNC_TO_LDS_B128
//   (4 b128 issues per wave per stage), double buffered against compute.
// ---------------------------------------------------------------------------
__global__ __launch_bounds__(128)
void flash_attn_kernel(const __bf16* __restrict__ theta,  // [8][4096][64]
                       const __bf16* __restrict__ phi,    // [8][1024][64]
                       const __bf16* __restrict__ g_t,    // [8][64][1024]
                       __bf16*       __restrict__ attng)  // [8][4096][64]
{
  __shared__ __bf16 kphi[2][32][64];   // [buf][key][c]   4KB each
  __shared__ __bf16 kg[2][64][32];     // [buf][c][key]   4KB each
  __shared__ __bf16 plds[4][16][32];   // per-wave P staging

  const int tid  = threadIdx.x;
  const int wave = tid >> 5;
  const int lane = tid & 31;
  const int tile = blockIdx.x * 4 + wave;   // 0..2047 (block stays in one batch)
  const int n  = tile >> 8;
  const int q0 = (tile & 255) * 16;

  const __bf16* th = theta + (long)n * 4096 * 64;
  const __bf16* ph = phi   + (long)n * 1024 * 64;
  const __bf16* gt = g_t   + (long)n * 64 * 1024;

  const int r      = lane & 15;
  const int hiofs  = (lane & 16) ? 8 : 0;   // A-layout K-segment shift
  const int hiofsB = (lane & 16) ? 16 : 0;  // K-group shift for contiguous k order (unused)
  const int rowofs = (lane & 16) ? 8 : 0;   // C-layout row group

  // ---- cooperative async stage of one 32-key chunk (4 b128 per wave) ----
  auto stage = [&](int buf, int k0) {
#pragma unroll
    for (int i = 0; i < 2; ++i) {           // phi: 256 units of 8 bf16
      int u = i * 128 + tid;
      int key = u >> 3, co = (u & 7) * 8;
      async_copy16(&kphi[buf][key][co], ph + (long)(k0 + key) * 64 + co);
    }
#pragma unroll
    for (int i = 0; i < 2; ++i) {           // g^T: 256 units of 8 bf16
      int u = i * 128 + tid;
      int c = u >> 2, ko = (u & 3) * 8;
      async_copy16(&kg[buf][c][ko], gt + (long)c * 1024 + k0 + ko);
    }
  };

  // Q tile 16x64 in A layout (2 k-steps)
  v16bf qa[2];
#pragma unroll
  for (int kt = 0; kt < 2; ++kt)
    qa[kt] = load_a_bf16(th + (long)(q0 + r) * 64 + kt * 32 + hiofs);

  v8f O[4];
#pragma unroll
  for (int nt = 0; nt < 4; ++nt)
#pragma unroll
    for (int v = 0; v < 8; ++v) O[nt][v] = 0.0f;

  float m_r[8], l_r[8];
#pragma unroll
  for (int v = 0; v < 8; ++v) { m_r[v] = -1.0e30f; l_r[v] = 0.0f; }

  stage(0, 0);
  stage(1, 32);

  for (int ci = 0; ci < 32; ++ci) {
    const int buf = ci & 1;
    // stage ci complete when only the next stage's 4 asyncs remain (in-order)
    if (ci < 31) wait_async_le4(); else wait_async_0();
    __syncthreads();

    // ---- S = Q . K^T  (two 16-key column tiles), B from LDS ----
    v8f S0, S1;
#pragma unroll
    for (int v = 0; v < 8; ++v) { S0[v] = 0.0f; S1[v] = 0.0f; }
#pragma unroll
    for (int kt = 0; kt < 2; ++kt) {
      S0 = WMMA_BF16(qa[kt], load_a_bf16(&kphi[buf][r][kt * 32 + hiofs]),      S0);
      S1 = WMMA_BF16(qa[kt], load_a_bf16(&kphi[buf][16 + r][kt * 32 + hiofs]), S1);
    }

    // ---- online softmax (row stats per lane: rows v+rowofs) ----
    float p0[8], p1[8];
#pragma unroll
    for (int v = 0; v < 8; ++v) {
      float mx   = halfmax(fmaxf(S0[v], S1[v]));
      float mnew = fmaxf(m_r[v], mx);
      float sc   = __expf(m_r[v] - mnew);
      m_r[v] = mnew;
      p0[v] = __expf(S0[v] - mnew);
      p1[v] = __expf(S1[v] - mnew);
      float rs = halfsum(p0[v] + p1[v]);
      l_r[v] = l_r[v] * sc + rs;
#pragma unroll
      for (int nt = 0; nt < 4; ++nt) O[nt][v] *= sc;
    }

    // ---- P: C-layout -> A-layout via LDS (wave-private buffer) ----
#pragma unroll
    for (int v = 0; v < 8; ++v) {
      int row = v + rowofs;
      plds[wave][row][r]      = (__bf16)p0[v];
      plds[wave][row][16 + r] = (__bf16)p1[v];
    }
    v16bf pa = load_a_bf16(&plds[wave][r][hiofs]);

    // ---- O += P . G   (g^T tile from LDS, contiguous per lane) ----
#pragma unroll
    for (int nt = 0; nt < 4; ++nt)
      O[nt] = WMMA_BF16(pa, load_a_bf16(&kg[buf][nt * 16 + r][hiofs]), O[nt]);

    __syncthreads();                         // everyone done reading buf
    if (ci + 2 < 32) stage(buf, (ci + 2) * 32);
  }

  // ---- normalize and emit attn_g (bf16, feeds final conv) ----
#pragma unroll
  for (int v = 0; v < 8; ++v) {
    float inv = 1.0f / l_r[v];
    long row = (long)n * 4096 + q0 + v + rowofs;
#pragma unroll
    for (int nt = 0; nt < 4; ++nt)
      attng[row * 64 + nt * 16 + r] = (__bf16)(O[nt][v] * inv);
  }
}

// ---------------------------------------------------------------------------
// Host-side launch sequence (graph-capture safe: only kernel launches).
// Workspace layout (bytes):
//   [0,4M) theta | [4M,8M) tmp | [8M,9M) phi | [9M,10M) g_t |
//   [10M,14M) attn_g | [14M,..) packed weights 4 x 8KB
// ---------------------------------------------------------------------------
extern "C" void kernel_launch(void* const* d_in, const int* in_sizes, int n_in,
                              void* d_out, int out_size, void* d_ws, size_t ws_size,
                              hipStream_t stream) {
  const float* x       = (const float*)d_in[0];
  const float* w_theta = (const float*)d_in[1];
  const float* b_theta = (const float*)d_in[2];
  const float* w_phi   = (const float*)d_in[3];
  const float* b_phi   = (const float*)d_in[4];
  const float* w_g     = (const float*)d_in[5];
  const float* b_g     = (const float*)d_in[6];
  const float* w_ag    = (const float*)d_in[7];
  const float* b_ag    = (const float*)d_in[8];
  const float* sigma   = (const float*)d_in[9];

  char* ws = (char*)d_ws;
  __bf16* theta = (__bf16*)(ws);
  __bf16* tmp   = (__bf16*)(ws + (4l  << 20));
  __bf16* phi   = (__bf16*)(ws + (8l  << 20));
  __bf16* g_t   = (__bf16*)(ws + (9l  << 20));
  __bf16* attng = (__bf16*)(ws + (10l << 20));
  __bf16* wp_th = (__bf16*)(ws + (14l << 20));
  __bf16* wp_ph = wp_th + 4096;
  __bf16* wp_g  = wp_ph + 4096;
  __bf16* wp_ag = wp_g  + 4096;

  pack_weight_kernel<<<1, 256, 0, stream>>>(w_theta, wp_th);
  pack_weight_kernel<<<1, 256, 0, stream>>>(w_phi,   wp_ph);
  pack_weight_kernel<<<1, 256, 0, stream>>>(w_g,     wp_g);
  pack_weight_kernel<<<1, 256, 0, stream>>>(w_ag,    wp_ag);

  const int M = 8 * 64 * 64;                 // 32768 pixel rows
  const dim3 gBlk(128);
  const dim3 gGrd(M / 64);

  conv1x1_wmma_kernel<<<gGrd, gBlk, 0, stream>>>(
      x, nullptr, wp_th, b_theta, theta, nullptr, nullptr, nullptr, M);
  conv1x1_wmma_kernel<<<gGrd, gBlk, 0, stream>>>(
      x, nullptr, wp_ph, b_phi, tmp, nullptr, nullptr, nullptr, M);
  pool_kernel<<<2048, 256, 0, stream>>>(tmp, phi, 0);
  conv1x1_wmma_kernel<<<gGrd, gBlk, 0, stream>>>(
      x, nullptr, wp_g, b_g, tmp, nullptr, nullptr, nullptr, M);
  pool_kernel<<<2048, 256, 0, stream>>>(tmp, g_t, 1);

  flash_attn_kernel<<<512, 128, 0, stream>>>(theta, phi, g_t, attng);

  conv1x1_wmma_kernel<<<gGrd, gBlk, 0, stream>>>(
      nullptr, attng, wp_ag, b_ag, nullptr, (float*)d_out, x, sigma, M);
}
